// RelativePositionSelfAttention_15908558864431
// MI455X (gfx1250) — matrix operand order repaired and checked by reading, not compiled, not run on hardware
//
#include <hip/hip_runtime.h>
#include <hip/hip_bf16.h>

typedef __attribute__((ext_vector_type(16))) _Float16 v16h;
typedef __attribute__((ext_vector_type(8)))  _Float16 v8h;
typedef __attribute__((ext_vector_type(4)))  _Float16 v4h;
typedef __attribute__((ext_vector_type(8)))  float    v8f;
typedef __attribute__((ext_vector_type(4)))  float    v4f;

#define E_DIM   1024
#define H_NUM   16
#define D_DIM   64
#define S_LEN   2048
#define B_NUM   4
#define BS_ROWS (B_NUM * S_LEN)   /* 8192 */
#define QK_SCALE 0.125f           /* 1/sqrt(64) */

static __device__ __forceinline__ v8f wmma_f16(v16h a, v16h b, v8f c) {
    // v_wmma_f32_16x16x32_f16: D = A(16x32 f16) x B(32x16 f16) + C(16x16 f32)
    return __builtin_amdgcn_wmma_f32_16x16x32_f16(
        /*neg_a=*/false, a, /*neg_b=*/false, b,
        /*c_mod=*/(short)0, c, /*reuse_a=*/false, /*reuse_b=*/false);
}

// ---------------------------------------------------------------------------
// fp32 -> f16 conversion (vectorized 4 elems/thread)
// ---------------------------------------------------------------------------
__global__ void cvt_f32_f16_kernel(const float* __restrict__ src,
                                   _Float16* __restrict__ dst, int n) {
    int i = (blockIdx.x * blockDim.x + threadIdx.x) * 4;
    if (i + 3 < n) {
        v4f v = *(const v4f*)(src + i);
        v4h h;
        h.x = (_Float16)v.x; h.y = (_Float16)v.y;
        h.z = (_Float16)v.z; h.w = (_Float16)v.w;
        *(v4h*)(dst + i) = h;
    } else {
        for (int j = i; j < n; ++j) dst[j] = (_Float16)src[j];
    }
}

// ---------------------------------------------------------------------------
// QKV projection: qkv = x @ W1^T + b1 ; scatter into Q(scaled), K, V^T (f16)
// One wave -> 16 rows x 64 cols (4 WMMA accumulators), K-loop over E in 32s.
// A-frag: lane holds row (l16), elements 0..7 at k=kb+half*16, 8..15 at
// k=kb+8+half*16 (two 16B loads). B-frag: lane holds col n=l16 -> row n of W1
// (torch Linear: y = x @ W^T), contiguous 16 halves at kb+half*16 (one 32B load).
// ---------------------------------------------------------------------------
__global__ void __launch_bounds__(128) qkv_gemm_kernel(
    const _Float16* __restrict__ xh, const _Float16* __restrict__ w1,
    const float* __restrict__ b1,
    _Float16* __restrict__ qh, _Float16* __restrict__ kh,
    _Float16* __restrict__ vT)
{
    const int wave = threadIdx.x >> 5;
    const int lane = threadIdx.x & 31;
    const int hsel = lane >> 4;
    const int l16  = lane & 15;
    const int gw   = blockIdx.x * 4 + wave;
    const int NCOLG = (3 * E_DIM) / 64;     // 48 column-groups of 64
    const int rowt = gw / NCOLG;            // 0..511
    const int colg = gw % NCOLG;
    const int nbase = colg * 64;
    const int row0  = rowt * 16;

    const _Float16* arow = xh + (size_t)(row0 + l16) * E_DIM;

    v8f acc[4] = {{}, {}, {}, {}};

    for (int kb = 0; kb < E_DIM; kb += 32) {
        v8h alo = *(const v8h*)(arow + kb + hsel * 16);
        v8h ahi = *(const v8h*)(arow + kb + 8 + hsel * 16);
        v16h a;
        #pragma unroll
        for (int i = 0; i < 8; ++i) { a[i] = alo[i]; a[8 + i] = ahi[i]; }
        #pragma unroll
        for (int t = 0; t < 4; ++t) {
            const _Float16* bp =
                w1 + (size_t)(nbase + t * 16 + l16) * E_DIM + kb + hsel * 16;
            v16h b = *(const v16h*)bp;
            acc[t] = wmma_f16(a, b, acc[t]);
        }
    }

    // Epilogue: C layout row = hsel*8+r, col = l16 within tile.
    #pragma unroll
    for (int t = 0; t < 4; ++t) {
        const int col    = nbase + t * 16 + l16;
        const int region = col >> 10;           // 0=q 1=k 2=v (uniform per tile)
        const int e      = col & (E_DIM - 1);
        const int hh     = e >> 6;
        const int dd     = e & (D_DIM - 1);
        const float bv   = b1[col];
        #pragma unroll
        for (int r = 0; r < 8; ++r) {
            const int rr = row0 + hsel * 8 + r;     // row in [0, BS)
            const int bb = rr >> 11;                // / S
            const int ss = rr & (S_LEN - 1);
            const int bh = bb * H_NUM + hh;
            const float val = acc[t][r] + bv;
            if (region == 0) {
                qh[((size_t)bh * S_LEN + ss) * D_DIM + dd] =
                    (_Float16)(val * QK_SCALE);
            } else if (region == 1) {
                kh[((size_t)bh * S_LEN + ss) * D_DIM + dd] = (_Float16)val;
            } else {
                vT[((size_t)bh * D_DIM + dd) * S_LEN + ss] = (_Float16)val;
            }
        }
    }
}

// ---------------------------------------------------------------------------
// Flash attention with rel-pos bias. One wave owns one (b,h) x 16-query tile,
// streams keys 32 at a time: 4 WMMA (scores) + online softmax + 4 WMMA (P@V).
// P is relayed C-layout -> A-layout through LDS (wave-private 1KB slab).
// ---------------------------------------------------------------------------
__global__ void __launch_bounds__(128) attn_flash_kernel(
    const _Float16* __restrict__ qh, const _Float16* __restrict__ kh,
    const _Float16* __restrict__ vT, const float* __restrict__ rel_pos,
    _Float16* __restrict__ ao)
{
    __shared__ _Float16 pbuf[4][16 * 32];

    const int wave = threadIdx.x >> 5;
    const int lane = threadIdx.x & 31;
    const int hsel = lane >> 4;
    const int l16  = lane & 15;
    const int gw   = blockIdx.x * 4 + wave;
    const int bh   = gw >> 7;                // 0..63 (b*H+h)
    const int qt   = gw & 127;
    const int qbase = qt * 16;
    const int bb   = bh >> 4;
    const int hh   = bh & 15;

    // Q A-fragments for kb=0 and kb=32 (held in registers for whole loop)
    const _Float16* qrow = qh + ((size_t)bh * S_LEN + qbase + l16) * D_DIM;
    v16h aq0, aq1;
    {
        v8h lo0 = *(const v8h*)(qrow + hsel * 16);
        v8h hi0 = *(const v8h*)(qrow + 8 + hsel * 16);
        v8h lo1 = *(const v8h*)(qrow + 32 + hsel * 16);
        v8h hi1 = *(const v8h*)(qrow + 40 + hsel * 16);
        #pragma unroll
        for (int i = 0; i < 8; ++i) {
            aq0[i] = lo0[i]; aq0[8 + i] = hi0[i];
            aq1[i] = lo1[i]; aq1[8 + i] = hi1[i];
        }
    }

    v8f o[4] = {{}, {}, {}, {}};
    float m_s[8], l_s[8];
    #pragma unroll
    for (int r = 0; r < 8; ++r) { m_s[r] = -1e30f; l_s[r] = 0.0f; }

    _Float16* pw = &pbuf[wave][0];

    for (int jb = 0; jb < S_LEN; jb += 32) {
        // ---- scores: S = Q @ K^T (two 16x16 tiles, keys jb..jb+31) ----
        v8f s0 = {}, s1 = {};
        {
            const _Float16* k0 = kh + ((size_t)bh * S_LEN + jb + l16) * D_DIM;
            v16h b0 = *(const v16h*)(k0 + hsel * 16);
            v16h b1 = *(const v16h*)(k0 + 32 + hsel * 16);
            s0 = wmma_f16(aq0, b0, s0);
            s0 = wmma_f16(aq1, b1, s0);
            const _Float16* k1 = kh + ((size_t)bh * S_LEN + jb + 16 + l16) * D_DIM;
            v16h b2 = *(const v16h*)(k1 + hsel * 16);
            v16h b3 = *(const v16h*)(k1 + 32 + hsel * 16);
            s1 = wmma_f16(aq0, b2, s1);
            s1 = wmma_f16(aq1, b3, s1);
        }
        // ---- additive rel-pos bias (L2-resident, reused by all 64 bh) ----
        #pragma unroll
        for (int r = 0; r < 8; ++r) {
            const int qr = qbase + hsel * 8 + r;
            s0[r] += rel_pos[(size_t)qr * S_LEN + jb + l16];
            s1[r] += rel_pos[(size_t)qr * S_LEN + jb + 16 + l16];
        }
        // ---- online softmax (rows live in 16-lane groups) ----
        float p0[8], p1[8], alpha[8];
        #pragma unroll
        for (int r = 0; r < 8; ++r) {
            float mx = fmaxf(s0[r], s1[r]);
            #pragma unroll
            for (int off = 1; off < 16; off <<= 1)
                mx = fmaxf(mx, __shfl_xor(mx, off, 32));
            const float mnew = fmaxf(m_s[r], mx);
            alpha[r] = __expf(m_s[r] - mnew);
            p0[r] = __expf(s0[r] - mnew);
            p1[r] = __expf(s1[r] - mnew);
            float rs = p0[r] + p1[r];
            #pragma unroll
            for (int off = 1; off < 16; off <<= 1)
                rs += __shfl_xor(rs, off, 32);
            l_s[r] = l_s[r] * alpha[r] + rs;
            m_s[r] = mnew;
        }
        #pragma unroll
        for (int t = 0; t < 4; ++t)
            #pragma unroll
            for (int r = 0; r < 8; ++r) o[t][r] *= alpha[r];

        // ---- stage P (16x32 f16) to LDS, C-layout -> A-layout relayout ----
        #pragma unroll
        for (int r = 0; r < 8; ++r) {
            pw[(hsel * 8 + r) * 32 + l16]      = (_Float16)p0[r];
            pw[(hsel * 8 + r) * 32 + 16 + l16] = (_Float16)p1[r];
        }
        asm volatile("s_wait_dscnt 0" ::: "memory");   // DS in-order per wave
        v16h pa;
        {
            v8h plo = *(const v8h*)(pw + l16 * 32 + hsel * 16);
            v8h phi = *(const v8h*)(pw + l16 * 32 + 8 + hsel * 16);
            #pragma unroll
            for (int i = 0; i < 8; ++i) { pa[i] = plo[i]; pa[8 + i] = phi[i]; }
        }
        // ---- O += P @ V (V^T layout makes B-frags one contiguous 32B load) ----
        #pragma unroll
        for (int t = 0; t < 4; ++t) {
            const _Float16* vp =
                vT + ((size_t)bh * D_DIM + t * 16 + l16) * S_LEN + jb + hsel * 16;
            v16h vb = *(const v16h*)vp;
            o[t] = wmma_f16(pa, vb, o[t]);
        }
        asm volatile("s_wait_dscnt 0" ::: "memory");   // don't let next stores pass loads
    }

    // ---- normalize and scatter to [B,S,E] f16 for out-proj ----
    #pragma unroll
    for (int t = 0; t < 4; ++t) {
        #pragma unroll
        for (int r = 0; r < 8; ++r) {
            const int qr = qbase + hsel * 8 + r;
            const float val = o[t][r] / l_s[r];
            ao[((size_t)bb * S_LEN + qr) * E_DIM + hh * D_DIM + t * 16 + l16] =
                (_Float16)val;
        }
    }
}

// ---------------------------------------------------------------------------
// Output projection: out = ao @ W2^T + b2  (f32 output)
// ---------------------------------------------------------------------------
__global__ void __launch_bounds__(128) out_gemm_kernel(
    const _Float16* __restrict__ ah, const _Float16* __restrict__ w2,
    const float* __restrict__ b2, float* __restrict__ out)
{
    const int wave = threadIdx.x >> 5;
    const int lane = threadIdx.x & 31;
    const int hsel = lane >> 4;
    const int l16  = lane & 15;
    const int gw   = blockIdx.x * 4 + wave;
    const int NCOLG = E_DIM / 64;            // 16
    const int rowt = gw / NCOLG;
    const int colg = gw % NCOLG;
    const int nbase = colg * 64;
    const int row0  = rowt * 16;

    const _Float16* arow = ah + (size_t)(row0 + l16) * E_DIM;

    v8f acc[4] = {{}, {}, {}, {}};

    for (int kb = 0; kb < E_DIM; kb += 32) {
        v8h alo = *(const v8h*)(arow + kb + hsel * 16);
        v8h ahi = *(const v8h*)(arow + kb + 8 + hsel * 16);
        v16h a;
        #pragma unroll
        for (int i = 0; i < 8; ++i) { a[i] = alo[i]; a[8 + i] = ahi[i]; }
        #pragma unroll
        for (int t = 0; t < 4; ++t) {
            const _Float16* bp =
                w2 + (size_t)(nbase + t * 16 + l16) * E_DIM + kb + hsel * 16;
            v16h b = *(const v16h*)bp;
            acc[t] = wmma_f16(a, b, acc[t]);
        }
    }

    #pragma unroll
    for (int t = 0; t < 4; ++t) {
        const int col  = nbase + t * 16 + l16;
        const float bv = b2[col];
        #pragma unroll
        for (int r = 0; r < 8; ++r) {
            const int rr = row0 + hsel * 8 + r;
            out[(size_t)rr * E_DIM + col] = acc[t][r] + bv;
        }
    }
}

// ---------------------------------------------------------------------------
extern "C" void kernel_launch(void* const* d_in, const int* in_sizes, int n_in,
                              void* d_out, int out_size, void* d_ws, size_t ws_size,
                              hipStream_t stream)
{
    (void)in_sizes; (void)n_in; (void)out_size; (void)ws_size;
    const float* x    = (const float*)d_in[0];
    const float* w1f  = (const float*)d_in[1];
    const float* b1   = (const float*)d_in[2];
    const float* w2f  = (const float*)d_in[3];
    const float* b2   = (const float*)d_in[4];
    const float* rel  = (const float*)d_in[5];
    float* out        = (float*)d_out;

    // Workspace partition (~88 MB total, 256B aligned chunks)
    char* ws = (char*)d_ws;
    size_t off = 0;
    auto take = [&](size_t bytes) {
        char* p = ws + off;
        off += (bytes + 255) & ~(size_t)255;
        return p;
    };
    _Float16* xh  = (_Float16*)take((size_t)BS_ROWS * E_DIM * 2);       // 16 MB
    _Float16* w1h = (_Float16*)take((size_t)3 * E_DIM * E_DIM * 2);     //  6 MB
    _Float16* w2h = (_Float16*)take((size_t)E_DIM * E_DIM * 2);         //  2 MB
    _Float16* qh  = (_Float16*)take((size_t)BS_ROWS * E_DIM * 2);       // 16 MB
    _Float16* kh  = (_Float16*)take((size_t)BS_ROWS * E_DIM * 2);       // 16 MB
    _Float16* vTh = (_Float16*)take((size_t)BS_ROWS * E_DIM * 2);       // 16 MB
    _Float16* aoh = (_Float16*)take((size_t)BS_ROWS * E_DIM * 2);       // 16 MB

    const int nx = BS_ROWS * E_DIM;           // 8,388,608
    const int n1 = 3 * E_DIM * E_DIM;         // 3,145,728
    const int n2 = E_DIM * E_DIM;             // 1,048,576
    cvt_f32_f16_kernel<<<(nx / 4 + 255) / 256, 256, 0, stream>>>(x, xh, nx);
    cvt_f32_f16_kernel<<<(n1 / 4 + 255) / 256, 256, 0, stream>>>(w1f, w1h, n1);
    cvt_f32_f16_kernel<<<(n2 / 4 + 255) / 256, 256, 0, stream>>>(w2f, w2h, n2);

    // 512 row-tiles x 48 col-groups = 24576 waves / 4 per block
    qkv_gemm_kernel<<<6144, 128, 0, stream>>>(xh, w1h, b1, qh, kh, vTh);

    // 64 (b,h) x 128 q-tiles = 8192 waves / 4 per block
    attn_flash_kernel<<<2048, 128, 0, stream>>>(qh, kh, vTh, rel, aoh);

    // 512 row-tiles x 16 col-groups = 8192 waves / 4 per block
    out_gemm_kernel<<<2048, 128, 0, stream>>>(aoh, w2h, b2, out);
}
